// MultiHeadSelfAttention_69286412419614
// MI455X (gfx1250) — compile-verified, matrix operand order checked by reading, not run on hardware
//
#include <hip/hip_runtime.h>
#include <hip/hip_bf16.h>
#include <stddef.h>

// ---------------------------------------------------------------------------
// MultiHeadSelfAttention for MI455X (gfx1250, wave32, WMMA bf16 16x16x32).
// B=2, S=2048, E=1024, H=16, D=64.  All matmuls run on v_wmma_f32_16x16x32_bf16.
// ---------------------------------------------------------------------------

#define BATCH 2
#define SEQ   2048
#define EMB   1024
#define HEADS 16
#define HDIM  64
#define MROWS (BATCH * SEQ)   // 4096

typedef __attribute__((ext_vector_type(16))) __bf16 v16bf;
typedef __attribute__((ext_vector_type(8)))  __bf16 v8bf;
typedef __attribute__((ext_vector_type(8)))  float  v8f;

__device__ __forceinline__ unsigned short f32_to_bf16(float f) {
  unsigned int u = __float_as_uint(f);
  u += 0x7fffu + ((u >> 16) & 1u);   // round-to-nearest-even
  return (unsigned short)(u >> 16);
}

__device__ __forceinline__ v8bf ld8(const unsigned short* p) {
  return *reinterpret_cast<const v8bf*>(p);
}

// Assemble a 16-element fragment from two contiguous 16-byte chunks.
// Chunk0 -> VGPRs 0..3 (elems 0..7), chunk1 -> VGPRs 4..7 (elems 8..15),
// matching the ISA 16-bit A/B fragment layout (K 0-7/16-23 for lanes 0-15,
// K 8-15/24-31 for lanes 16-31, handled by caller via +8*half addressing).
__device__ __forceinline__ v16bf frag2(const unsigned short* p0,
                                       const unsigned short* p1) {
  v8bf a = ld8(p0), b = ld8(p1);
  v16bf f;
#pragma unroll
  for (int i = 0; i < 8; ++i) { f[i] = a[i]; f[i + 8] = b[i]; }
  return f;
}

__device__ __forceinline__ v8f wmma_bf16(v16bf a, v16bf b, v8f c) {
  return __builtin_amdgcn_wmma_f32_16x16x32_bf16(false, a, false, b,
                                                 (short)0, c, false, false);
}

// ---------------------------------------------------------------------------
// Kernel 1: f32 -> bf16 elementwise convert (x).
// ---------------------------------------------------------------------------
__global__ void conv_bf16_kernel(const float* __restrict__ X,
                                 unsigned short* __restrict__ Xb, int n) {
  int i = blockIdx.x * blockDim.x + threadIdx.x;
  if (i < n) Xb[i] = f32_to_bf16(X[i]);
}

// ---------------------------------------------------------------------------
// Kernel 2: W (E x E f32 row-major) -> bf16 column-major  WT[n*E + k].
// Reads coalesced; writes hit L2 (tiny: 2 MB per weight).
// ---------------------------------------------------------------------------
__global__ void transpose_bf16_kernel(const float* __restrict__ W,
                                      unsigned short* __restrict__ WT) {
  int idx = blockIdx.x * blockDim.x + threadIdx.x;   // 0 .. E*E-1
  int n = idx & (EMB - 1);
  int k = idx >> 10;
  WT[(size_t)n * EMB + k] = f32_to_bf16(W[idx]);
}

// ---------------------------------------------------------------------------
// Kernel 3: bf16 GEMM  C(MxN) = A(MxK,row) * B(KxN, stored col-major Bt[n*K+k])
// One wave computes a 32x64 output tile (2x4 C frags), K-loop step 32.
// Each B-fragment is reused for two WMMAs (24 B loaded per WMMA vs 40 B for
// a 16x64 tile) -- better L2-bandwidth/compute balance on the big GEMMs.
// MODE 0: f32 row-major out + bias (final projection)
// MODE 1: bf16 out in (B,H,S,D)  (Q and K)
// MODE 2: bf16 out in (B,H,D,S)  (V transposed for attention B-fragments)
// ---------------------------------------------------------------------------
template <int MODE>
__global__ void __launch_bounds__(128)
gemm_bf16_kernel(const unsigned short* __restrict__ A,
                 const unsigned short* __restrict__ Bt,
                 const float* __restrict__ bias,
                 void* __restrict__ Out,
                 int M, int N, int K) {
  int wave = (blockIdx.x * blockDim.x + threadIdx.x) >> 5;
  int lane = threadIdx.x & 31;
  int r  = lane & 15;
  int hf = lane >> 4;            // half-wave select
  int mtiles = M >> 5;           // 32-row tiles
  int mt = wave % mtiles;
  int ng = wave / mtiles;
  int mbase = mt << 5;
  int nbase = ng << 6;
  if (nbase >= N) return;

  v8f acc[2][4];
#pragma unroll
  for (int t = 0; t < 2; ++t)
#pragma unroll
    for (int f = 0; f < 4; ++f) acc[t][f] = (v8f){};

  const unsigned short* arow0 = A + (size_t)(mbase + r) * K;
  const unsigned short* arow1 = A + (size_t)(mbase + 16 + r) * K;
  for (int k0 = 0; k0 < K; k0 += 32) {
    v16bf aF0 = frag2(arow0 + k0 + 8 * hf, arow0 + k0 + 16 + 8 * hf);
    v16bf aF1 = frag2(arow1 + k0 + 8 * hf, arow1 + k0 + 16 + 8 * hf);
#pragma unroll
    for (int f = 0; f < 4; ++f) {
      const unsigned short* bcol =
          Bt + (size_t)(nbase + f * 16 + r) * K + k0 + 8 * hf;
      v16bf bF = frag2(bcol, bcol + 16);
      acc[0][f] = wmma_bf16(aF0, bF, acc[0][f]);
      acc[1][f] = wmma_bf16(aF1, bF, acc[1][f]);
    }
  }

#pragma unroll
  for (int t = 0; t < 2; ++t) {
#pragma unroll
    for (int f = 0; f < 4; ++f) {
#pragma unroll
      for (int j = 0; j < 8; ++j) {
        int row = mbase + t * 16 + j + 8 * hf;  // C/D layout: lanes16-31 -> M+8
        int col = nbase + f * 16 + r;
        float v = acc[t][f][j] + bias[col];
        if (MODE == 0) {
          ((float*)Out)[(size_t)row * N + col] = v;
        } else {
          int b = row / SEQ, s = row % SEQ;
          int h = col >> 6, d = col & 63;
          size_t idx;
          if (MODE == 1)
            idx = (((size_t)(b * HEADS + h)) * SEQ + s) * HDIM + d;
          else
            idx = (((size_t)(b * HEADS + h)) * HDIM + d) * SEQ + s;
          ((unsigned short*)Out)[idx] = f32_to_bf16(v);
        }
      }
    }
  }
}

// ---------------------------------------------------------------------------
// Kernel 4: causal flash attention.  One wave per (b, h, 16-query tile).
// Scores S = Q K^T via 4 WMMAs per 32-key block; online softmax with
// cross-lane reductions over the 16-lane half-groups holding each C-row;
// P re-shaped C-layout -> A-layout through a per-wave LDS tile; P*V via
// 4 more WMMAs.  Mask follows reference exactly: masked score = -10000.
// ---------------------------------------------------------------------------
__global__ void __launch_bounds__(128)
attn_kernel(const unsigned short* __restrict__ Q,
            const unsigned short* __restrict__ Kb,
            const unsigned short* __restrict__ Vt,
            unsigned short* __restrict__ O) {
  __shared__ __align__(16) unsigned short ldsP[4][16][32];

  int gw = (blockIdx.x * blockDim.x + threadIdx.x) >> 5;
  int wl = threadIdx.x >> 5;
  int lane = threadIdx.x & 31;
  int r  = lane & 15;
  int hf = lane >> 4;

  int qtiles = SEQ >> 4;
  int qt = gw % qtiles;
  int bh = gw / qtiles;
  int qbase = qt << 4;

  // Q fragments (16 rows x 64 d, as two k=32 A-frags), loaded once.
  const unsigned short* qrow = Q + ((size_t)bh * SEQ + qbase + r) * HDIM;
  v16bf qf0 = frag2(qrow + 8 * hf,      qrow + 16 + 8 * hf);
  v16bf qf1 = frag2(qrow + 32 + 8 * hf, qrow + 48 + 8 * hf);

  v8f oA[4];
#pragma unroll
  for (int f = 0; f < 4; ++f) oA[f] = (v8f){};
  float mst[8], lst[8];
#pragma unroll
  for (int j = 0; j < 8; ++j) { mst[j] = -3.0e38f; lst[j] = 0.f; }

  const float scale = 0.125f;  // 1/sqrt(64)
  int nkb = (qbase + 16 + 31) >> 5;   // 32-key blocks up to causal frontier

  for (int kbi = 0; kbi < nkb; ++kbi) {
    int kbase = kbi << 5;

    // ---- scores: two 16-key column tiles, accumulate over d = 0..63 ----
    v8f s0 = (v8f){}, s1 = (v8f){};
    {
      const unsigned short* k0r = Kb + ((size_t)bh * SEQ + kbase + r) * HDIM;
      const unsigned short* k1r = Kb + ((size_t)bh * SEQ + kbase + 16 + r) * HDIM;
      v16bf k00 = frag2(k0r + 8 * hf,      k0r + 16 + 8 * hf);
      v16bf k01 = frag2(k0r + 32 + 8 * hf, k0r + 48 + 8 * hf);
      v16bf k10 = frag2(k1r + 8 * hf,      k1r + 16 + 8 * hf);
      v16bf k11 = frag2(k1r + 32 + 8 * hf, k1r + 48 + 8 * hf);
      s0 = wmma_bf16(qf0, k00, s0);
      s0 = wmma_bf16(qf1, k01, s0);
      s1 = wmma_bf16(qf0, k10, s1);
      s1 = wmma_bf16(qf1, k11, s1);
    }

    // ---- mask + online softmax statistics ----
    float p0[8], p1[8];
#pragma unroll
    for (int j = 0; j < 8; ++j) {
      int row = qbase + j + 8 * hf;
      int c0 = kbase + r, c1 = kbase + 16 + r;
      float a = (c0 <= row) ? s0[j] * scale : -10000.f;
      float b = (c1 <= row) ? s1[j] * scale : -10000.f;
      float mx = fmaxf(a, b);
      for (int off = 1; off < 16; off <<= 1)
        mx = fmaxf(mx, __shfl_xor(mx, off, 32));
      float mnew = fmaxf(mst[j], mx);
      float ea = __expf(a - mnew);
      float eb = __expf(b - mnew);
      float rs = ea + eb;
      for (int off = 1; off < 16; off <<= 1)
        rs += __shfl_xor(rs, off, 32);
      float corr = __expf(mst[j] - mnew);
      lst[j] = lst[j] * corr + rs;
      mst[j] = mnew;
#pragma unroll
      for (int f = 0; f < 4; ++f) oA[f][j] *= corr;
      p0[j] = ea;
      p1[j] = eb;
    }

    // ---- C-layout -> A-layout reshape of P through LDS ----
#pragma unroll
    for (int j = 0; j < 8; ++j) {
      ldsP[wl][j + 8 * hf][r]      = f32_to_bf16(p0[j]);
      ldsP[wl][j + 8 * hf][16 + r] = f32_to_bf16(p1[j]);
    }
    __asm volatile("s_wait_dscnt 0" ::: "memory");  // wave-local LDS RAW
    const unsigned short* prow = &ldsP[wl][r][0];
    v16bf pf = frag2(prow + 8 * hf, prow + 16 + 8 * hf);

    // ---- P (16x32) x V (32 x 64) via V^T contiguous B-fragments ----
#pragma unroll
    for (int f = 0; f < 4; ++f) {
      const unsigned short* vcol =
          Vt + ((size_t)bh * HDIM + f * 16 + r) * SEQ + kbase;
      v16bf vf = frag2(vcol + 8 * hf, vcol + 16 + 8 * hf);
      oA[f] = wmma_bf16(pf, vf, oA[f]);
    }
  }

  // ---- normalize, write O as (B, S, H*64+d) bf16 row-major ----
  int b = bh >> 4, h = bh & 15;
#pragma unroll
  for (int f = 0; f < 4; ++f) {
#pragma unroll
    for (int j = 0; j < 8; ++j) {
      int srow = qbase + j + 8 * hf;
      int e = h * HDIM + f * 16 + r;
      float v = oA[f][j] / lst[j];
      O[((size_t)b * SEQ + srow) * EMB + e] = f32_to_bf16(v);
    }
  }
}

// ---------------------------------------------------------------------------
// Host-side launch.  Workspace layout (40 MB used):
//   [ 0MB)  xbf (8MB)   -- reused as O bf16 after QKV GEMMs consume x
//   [ 8MB)  WqT (2MB)  [10MB) WkT  [12MB) WvT  [14MB) WoT
//   [16MB)  Q (8MB)    [24MB) K (8MB)    [32MB) V^T (8MB)
// ---------------------------------------------------------------------------
extern "C" void kernel_launch(void* const* d_in, const int* in_sizes, int n_in,
                              void* d_out, int out_size, void* d_ws,
                              size_t ws_size, hipStream_t stream) {
  const float* x  = (const float*)d_in[0];
  const float* Wq = (const float*)d_in[1];
  const float* bq = (const float*)d_in[2];
  const float* Wk = (const float*)d_in[3];
  const float* bk = (const float*)d_in[4];
  const float* Wv = (const float*)d_in[5];
  const float* bv = (const float*)d_in[6];
  const float* Wo = (const float*)d_in[7];
  const float* bo = (const float*)d_in[8];

  char* ws = (char*)d_ws;
  const size_t MB = 1u << 20;
  unsigned short* xbf = (unsigned short*)(ws + 0 * MB);
  unsigned short* Obf = xbf;  // alias: x dead after QKV projections
  unsigned short* WqT = (unsigned short*)(ws + 8 * MB);
  unsigned short* WkT = (unsigned short*)(ws + 10 * MB);
  unsigned short* WvT = (unsigned short*)(ws + 12 * MB);
  unsigned short* WoT = (unsigned short*)(ws + 14 * MB);
  unsigned short* Qb  = (unsigned short*)(ws + 16 * MB);
  unsigned short* Kbf = (unsigned short*)(ws + 24 * MB);
  unsigned short* Vt  = (unsigned short*)(ws + 32 * MB);

  // 1) x -> bf16
  {
    int n = MROWS * EMB;
    conv_bf16_kernel<<<(n + 255) / 256, 256, 0, stream>>>(x, xbf, n);
  }
  // 2) weights -> bf16 col-major
  {
    int blocks = (EMB * EMB) / 256;
    transpose_bf16_kernel<<<blocks, 256, 0, stream>>>(Wq, WqT);
    transpose_bf16_kernel<<<blocks, 256, 0, stream>>>(Wk, WkT);
    transpose_bf16_kernel<<<blocks, 256, 0, stream>>>(Wv, WvT);
    transpose_bf16_kernel<<<blocks, 256, 0, stream>>>(Wo, WoT);
  }
  // 3) Q/K/V projections: 2048 waves (32x64 tiles), 128 thr/block -> 512 blocks
  {
    int blocks = (MROWS / 32) * (EMB / 64) * 32 / 128;
    gemm_bf16_kernel<1><<<blocks, 128, 0, stream>>>(xbf, WqT, bq, Qb,
                                                    MROWS, EMB, EMB);
    gemm_bf16_kernel<1><<<blocks, 128, 0, stream>>>(xbf, WkT, bk, Kbf,
                                                    MROWS, EMB, EMB);
    gemm_bf16_kernel<2><<<blocks, 128, 0, stream>>>(xbf, WvT, bv, Vt,
                                                    MROWS, EMB, EMB);
  }
  // 4) attention: B*H*(S/16) = 4096 waves -> 1024 blocks of 128
  {
    int blocks = BATCH * HEADS * (SEQ / 16) * 32 / 128;
    attn_kernel<<<blocks, 128, 0, stream>>>(Qb, Kbf, Vt, Obf);
  }
  // 5) output projection -> f32 d_out
  {
    int blocks = (MROWS / 32) * (EMB / 64) * 32 / 128;
    gemm_bf16_kernel<0><<<blocks, 128, 0, stream>>>(Obf, WoT, bo, d_out,
                                                    MROWS, EMB, EMB);
  }
}
